// LSTM_79250736546388
// MI455X (gfx1250) — matrix operand order repaired and checked by reading, not compiled, not run on hardware
//
#include <hip/hip_runtime.h>

// ---------------------------------------------------------------------------
// Fused LSTM-scan for MI455X (gfx1250, wave32, WMMA).
//   B=256 T=2048 I=H=128.  16 WGs x 16 batch rows. Weight B-fragments are
//   register-resident (256 VGPRs/lane, gfx1250 VGPR-MSB); per step each of
//   8 waves issues 32x v_wmma_f32_16x16x32_bf16 fed by 16 ds_load_b128.
// ---------------------------------------------------------------------------

typedef __attribute__((ext_vector_type(16))) __bf16        v16bf;
typedef __attribute__((ext_vector_type(8)))  float         v8f;
typedef __attribute__((ext_vector_type(4)))  float         v4f;
typedef __attribute__((ext_vector_type(8)))  unsigned int  v8u;

#define B_SZ 256
#define T_SZ 2048
#define I_SZ 128
#define H_SZ 128
#define KDIM 256                 // I + H (concatenated A operand)
#define NDIM 512                 // 4 gates * H
#define ROWS 16                  // batch rows per workgroup
#define WS   264                 // Wt row stride (ushorts): 132 dwords -> bank spread
#define AS   264                 // A-tile row stride (ushorts)

// LDS layout (ushort units)
#define WT_OFF   0
#define AT_OFF   (NDIM * WS)                 // 512*264
#define BIAS_OFF (NDIM * WS + ROWS * AS)     // floats start here (byte-off divisible by 4)
#define SMEM_USHORTS (NDIM * WS + ROWS * AS + 2 * NDIM)   // + 512 floats of bias

__device__ __forceinline__ unsigned short f2bf(float f) {
    unsigned u = __builtin_bit_cast(unsigned, f);
    unsigned r = (u + 0x7FFFu + ((u >> 16) & 1u)) >> 16;   // round-to-nearest-even
    return (unsigned short)r;
}

__device__ __forceinline__ unsigned pack2bf(float lo, float hi) {
    return (unsigned)f2bf(lo) | ((unsigned)f2bf(hi) << 16);
}

// Load a 16-element bf16 fragment as two 16B LDS reads `gap` ushorts apart.
__device__ __forceinline__ v16bf lds_frag(const unsigned short* p, int gap) {
    uint4 a = *(const uint4*)(p);
    uint4 b = *(const uint4*)(p + gap);
    v8u u = { a.x, a.y, a.z, a.w, b.x, b.y, b.z, b.w };
    return __builtin_bit_cast(v16bf, u);
}

// sigmoid via v_exp_f32 + v_rcp_f32 (both TRANS ops -> co-execute with WMMA)
__device__ __forceinline__ float sigmoidf_fast(float v) {
    float e = __expf(-v);
    return __builtin_amdgcn_rcpf(1.0f + e);
}

__global__ __launch_bounds__(256, 1) void lstm_scan_wmma(
    const float* __restrict__ x,
    const float* __restrict__ Wfx, const float* __restrict__ Wfh,
    const float* __restrict__ Wix, const float* __restrict__ Wih,
    const float* __restrict__ Wox, const float* __restrict__ Woh,
    const float* __restrict__ Wcx, const float* __restrict__ Wch,
    const float* __restrict__ bf,  const float* __restrict__ bi,
    const float* __restrict__ bo,  const float* __restrict__ bc,
    float* __restrict__ out_h, float* __restrict__ out_c)
{
    extern __shared__ unsigned short smem[];
    unsigned short* Wt  = smem + WT_OFF;     // Wt[n][k] = W[k][n], bf16
    unsigned short* At  = smem + AT_OFF;     // A[m][k]: k<128 -> x_t, k>=128 -> h_{t-1}, bf16
    float* biasL        = (float*)(smem + BIAS_OFF);

    const int tid = threadIdx.x;
    const int b0  = blockIdx.x * ROWS;

    // ---------------- one-time: weights -> LDS (bf16, transposed, gate-blocked) ----------
    const float* WxA[4] = { Wfx, Wix, Wox, Wcx };
    const float* WhA[4] = { Wfh, Wih, Woh, Wch };
    for (int idx = tid; idx < NDIM * KDIM; idx += 256) {
        int n = idx >> 8, k = idx & 255;
        int g = n >> 7, col = n & 127;
        float v = (k < I_SZ) ? WxA[g][k * H_SZ + col]
                             : WhA[g][(k - I_SZ) * H_SZ + col];
        Wt[n * WS + k] = f2bf(v);
    }
    const float* biasA[4] = { bf, bi, bo, bc };
    for (int n = tid; n < NDIM; n += 256)
        biasL[n] = biasA[n >> 7][n & 127];

    // ---------------- init A-tile: x_0 and h = 0 ----------------
    const int xr = tid >> 4;            // 0..15  (batch row within tile)
    const int xc = (tid & 15) * 8;      // 0..120 (8 floats / 8 bf16 per thread)
    unsigned xoff = (unsigned)(b0 + xr) * (T_SZ * I_SZ) + (unsigned)xc;   // -> x_t slot
    {
        const v4f v0 = __builtin_nontemporal_load((const v4f*)(x + xoff));
        const v4f v1 = __builtin_nontemporal_load((const v4f*)(x + xoff + 4));
        uint4 pk;
        pk.x = pack2bf(v0.x, v0.y); pk.y = pack2bf(v0.z, v0.w);
        pk.z = pack2bf(v1.x, v1.y); pk.w = pack2bf(v1.z, v1.w);
        *(uint4*)(At + xr * AS + xc) = pk;
        uint4 z; z.x = 0; z.y = 0; z.z = 0; z.w = 0;
        *(uint4*)(At + xr * AS + I_SZ + xc) = z;
    }
    xoff += I_SZ;                        // now points at x_{t+1}
    __syncthreads();

    // ---------------- wave-local constants (ISA 7.12.2 fragment layouts) ----------------
    const int w  = tid >> 5;        // wave id: owns h-columns [16w, 16w+16)
    const int l  = tid & 31;
    const int nl = l & 15;          // N / M lane coordinate
    const int kh = l >> 4;          // K-half select

    const float bsf = biasL[0 * H_SZ + 16 * w + nl];
    const float bsi = biasL[1 * H_SZ + 16 * w + nl];
    const float bso = biasL[2 * H_SZ + 16 * w + nl];
    const float bsg = biasL[3 * H_SZ + 16 * w + nl];

    // A fragment base: row m = nl, K base = kh*8 (+kb*32); chunks at +0 and +16
    const unsigned short* Abase = At + nl * AS + kh * 8;

    // -------- hoist ALL weight B-fragments into registers (loop-invariant) --------
    // Bfrag[g][kb]: 16 bf16 per lane -> 4*8*8 = 256 VGPRs (gfx1250 VGPR-MSB path)
    v16bf Bfrag[4][8];
    #pragma unroll
    for (int g = 0; g < 4; ++g) {
        const unsigned short* Bb = Wt + (g * H_SZ + 16 * w + nl) * WS + kh * 8;
        #pragma unroll
        for (int kb = 0; kb < 8; ++kb)
            Bfrag[g][kb] = lds_frag(Bb + kb * 32, 16);
    }

    v8f creg = {};      // c[16x16] slice, f32, persistent in registers
    v8f hreg = {};      // h[16x16] slice

    const int colo = 16 * w + nl;   // global h-column this lane writes

    // persistent u32 element offsets for the 8 output rows (advance by H per step)
    unsigned offr[8];
    #pragma unroll
    for (int r = 0; r < 8; ++r)
        offr[r] = (unsigned)(b0 + r + kh * 8) * (T_SZ * H_SZ) + (unsigned)colo;

    for (int t = 0; t < T_SZ; ++t) {
        // -------- prefetch x_{t+1} into registers (completes under the matmul) --------
        v4f px0 = {0.f, 0.f, 0.f, 0.f}, px1 = {0.f, 0.f, 0.f, 0.f};
        if (t + 1 < T_SZ) {
            px0 = __builtin_nontemporal_load((const v4f*)(x + xoff));
            px1 = __builtin_nontemporal_load((const v4f*)(x + xoff + 4));
        }

        // -------- gates = bias + [x_t | h] @ [Wx ; Wh]  (K = 256, 32 WMMA) --------
        v8f af, ai, ao, ag;
        #pragma unroll
        for (int r = 0; r < 8; ++r) { af[r] = bsf; ai[r] = bsi; ao[r] = bso; ag[r] = bsg; }

        #pragma unroll
        for (int kb = 0; kb < 8; ++kb) {
            const v16bf a = lds_frag(Abase + kb * 32, 16);
            af = __builtin_amdgcn_wmma_f32_16x16x32_bf16(false, a, false, Bfrag[0][kb], (short)0, af, false, false);
            ai = __builtin_amdgcn_wmma_f32_16x16x32_bf16(false, a, false, Bfrag[1][kb], (short)0, ai, false, false);
            ao = __builtin_amdgcn_wmma_f32_16x16x32_bf16(false, a, false, Bfrag[2][kb], (short)0, ao, false, false);
            ag = __builtin_amdgcn_wmma_f32_16x16x32_bf16(false, a, false, Bfrag[3][kb], (short)0, ag, false, false);
        }

        // -------- elementwise cell update (wave-local, f32) --------
        #pragma unroll
        for (int r = 0; r < 8; ++r) {
            float f = sigmoidf_fast(af[r]);
            float i = sigmoidf_fast(ai[r]);
            float o = sigmoidf_fast(ao[r]);
            float g = ag[r];                      // linear candidate (faithful to source)
            float c = f * creg[r] + i * g;
            float h = (1.0f - o) * hreg[r] + o * c;
            creg[r] = c;
            hreg[r] = h;
        }

        // -------- stream h_t, c_t to HBM (write-once: non-temporal) --------
        #pragma unroll
        for (int r = 0; r < 8; ++r) {
            __builtin_nontemporal_store(hreg[r], out_h + offr[r]);
            __builtin_nontemporal_store(creg[r], out_c + offr[r]);
            offr[r] += H_SZ;
        }

        __syncthreads();   // everyone done reading this step's A-tile

        // -------- publish bf16 h into A-tile; install x_{t+1} --------
        #pragma unroll
        for (int r = 0; r < 8; ++r)
            At[(r + kh * 8) * AS + I_SZ + colo] = f2bf(hreg[r]);

        if (t + 1 < T_SZ) {
            uint4 pk;
            pk.x = pack2bf(px0.x, px0.y); pk.y = pack2bf(px0.z, px0.w);
            pk.z = pack2bf(px1.x, px1.y); pk.w = pack2bf(px1.z, px1.w);
            *(uint4*)(At + xr * AS + xc) = pk;
        }
        xoff += I_SZ;
        __syncthreads();   // A-tile ready for step t+1
    }
}

extern "C" void kernel_launch(void* const* d_in, const int* in_sizes, int n_in,
                              void* d_out, int out_size, void* d_ws, size_t ws_size,
                              hipStream_t stream) {
    const float* x   = (const float*)d_in[0];
    const float* Wfx = (const float*)d_in[1];
    const float* Wfh = (const float*)d_in[2];
    const float* Wix = (const float*)d_in[3];
    const float* Wih = (const float*)d_in[4];
    const float* Wox = (const float*)d_in[5];
    const float* Woh = (const float*)d_in[6];
    const float* Wcx = (const float*)d_in[7];
    const float* Wch = (const float*)d_in[8];
    const float* bf  = (const float*)d_in[9];
    const float* bi  = (const float*)d_in[10];
    const float* bo  = (const float*)d_in[11];
    const float* bc  = (const float*)d_in[12];

    float* out_h = (float*)d_out;
    float* out_c = out_h + (size_t)B_SZ * T_SZ * H_SZ;

    const size_t smem_bytes = (size_t)SMEM_USHORTS * sizeof(unsigned short); // ~274 KB < 320 KB/WGP
    hipLaunchKernelGGL(lstm_scan_wmma,
                       dim3(B_SZ / ROWS), dim3(256), smem_bytes, stream,
                       x, Wfx, Wfh, Wix, Wih, Wox, Woh, Wcx, Wch,
                       bf, bi, bo, bc, out_h, out_c);
}
